// ClipLoss_68229850464824
// MI455X (gfx1250) — matrix operand order repaired and checked by reading, not compiled, not run on hardware
//
#include <hip/hip_runtime.h>
#include <stdint.h>

#define BN 16384
#define DD 256
#define NT2 2                      // n-tiles (16 cols each) per iteration
#define TILE_ELEMS (NT2 * 16 * DD) // 8192 bf16 = 16 KB per stage
#define TILE_BYTES (TILE_ELEMS * 2)
#define NITER (BN / (16 * NT2))    // 512
#define NBUF 4                     // LDS ring buffers (fills issued 2 ahead)

typedef __attribute__((ext_vector_type(16))) __bf16 v16bf;
typedef __attribute__((ext_vector_type(8)))  float  v8f;

union Frag16 { v16bf v; uint4 q[2]; };

__device__ __forceinline__ uint16_t f2bf(float f) {
    uint32_t u = __builtin_bit_cast(uint32_t, f);
    uint32_t r = u + 0x7FFFu + ((u >> 16) & 1u);   // round-to-nearest-even
    return (uint16_t)(r >> 16);
}

// Async DMA of one 16 KB tile: 8 x b128 per lane, global -> LDS, ASYNCcnt.
// INST_OFFSET is applied to BOTH the LDS address (VDST) and the global
// address (VADDR); our LDS/global offsets advance in lockstep (2048 B).
__device__ __forceinline__ void async_fill_tile(uint32_t lds_addr, uint64_t gaddr) {
    asm volatile(
        "global_load_async_to_lds_b128 %0, %1, off\n\t"
        "global_load_async_to_lds_b128 %0, %1, off offset:2048\n\t"
        "global_load_async_to_lds_b128 %0, %1, off offset:4096\n\t"
        "global_load_async_to_lds_b128 %0, %1, off offset:6144\n\t"
        "global_load_async_to_lds_b128 %0, %1, off offset:8192\n\t"
        "global_load_async_to_lds_b128 %0, %1, off offset:10240\n\t"
        "global_load_async_to_lds_b128 %0, %1, off offset:12288\n\t"
        "global_load_async_to_lds_b128 %0, %1, off offset:14336"
        :: "v"(lds_addr), "v"(gaddr) : "memory");
}

// ASYNC ops complete in order: <=8 outstanding ==> previous tile retired.
__device__ __forceinline__ void wait_async_le8()  {
    asm volatile("s_wait_asynccnt 0x8" ::: "memory");
}
__device__ __forceinline__ void wait_async_zero() {
    asm volatile("s_wait_asynccnt 0x0" ::: "memory");
}

// ---------------------------------------------------------------------------
// 1) f32 -> bf16 conversion; logit_scale folded into M so both LSE sweeps see
//    identically scaled logits.
// ---------------------------------------------------------------------------
__global__ void k_convert(const float* __restrict__ M, const float* __restrict__ S,
                          const float* __restrict__ scalep,
                          uint16_t* __restrict__ Mbf, uint16_t* __restrict__ Sbf) {
    const float sc = *scalep;
    const int i = (blockIdx.x * blockDim.x + threadIdx.x) * 4;
    float4 m = *(const float4*)(M + i);
    float4 s = *(const float4*)(S + i);
    ushort4 mo, so;
    mo.x = f2bf(m.x * sc); mo.y = f2bf(m.y * sc);
    mo.z = f2bf(m.z * sc); mo.w = f2bf(m.w * sc);
    so.x = f2bf(s.x);      so.y = f2bf(s.y);
    so.z = f2bf(s.z);      so.w = f2bf(s.w);
    *(ushort4*)(Mbf + i) = mo;
    *(ushort4*)(Sbf + i) = so;
}

// ---------------------------------------------------------------------------
// 2) Exact f32 diagonal: diag[i] = scale * <m_i, s_i>. One wave32 per row.
// ---------------------------------------------------------------------------
__global__ void k_diag(const float* __restrict__ M, const float* __restrict__ S,
                       const float* __restrict__ scalep, float* __restrict__ diag) {
    const int wave = threadIdx.x >> 5;
    const int lane = threadIdx.x & 31;
    const int row  = blockIdx.x * 8 + wave;
    const float* mr = M + (size_t)row * DD;
    const float* sr = S + (size_t)row * DD;
    float acc = 0.f;
    #pragma unroll
    for (int k = lane; k < DD; k += 32) acc = fmaf(mr[k], sr[k], acc);
    #pragma unroll
    for (int msk = 16; msk >= 1; msk >>= 1) acc += __shfl_xor(acc, msk, 32);
    if (lane == 0) diag[row] = acc * (*scalep);
}

// ---------------------------------------------------------------------------
// 3) Row-wise online logsumexp of A @ B^T without materializing logits.
//    Block = 128 threads = 4 waves; wave w owns rows [blk*64 + w*16, +16).
//    A-panel (16x256 bf16) in registers as 8 WMMA A-fragments.
//    B tiles (32 cols x 256 K) in a 4-deep LDS ring, filled by
//    GLOBAL_LOAD_ASYNC_TO_LDS_B128 two tiles ahead (s_wait_asynccnt 8).
//    All 16 B fragments preloaded per iteration: ds-load burst -> WMMA burst.
// ---------------------------------------------------------------------------
__global__ __launch_bounds__(128) void k_lse(const uint16_t* __restrict__ A,
                                             const uint16_t* __restrict__ Bm,
                                             float* __restrict__ lse) {
    __shared__ __align__(16) uint16_t lds[NBUF][TILE_ELEMS];   // 4 x 16 KB

    const int tid  = threadIdx.x;
    const int wave = tid >> 5;
    const int lane = tid & 31;
    const int lo   = lane & 15;
    const int hi   = lane >> 4;            // 0 or 1
    const int row0 = blockIdx.x * 64 + wave * 16;

    // --- A fragments: 16-bit A-matrix 16x32 layout (ISA 7.12.2) ---
    // lane L: M = L&15; elems 0..7 -> K = k*32 + 8*hi + e ; elems 8..15 -> +16
    Frag16 afrag[8];
    {
        const uint16_t* arow = A + (size_t)(row0 + lo) * DD;
        #pragma unroll
        for (int k = 0; k < 8; ++k) {
            const int base = k * 32 + hi * 8;
            afrag[k].q[0] = *(const uint4*)(arow + base);
            afrag[k].q[1] = *(const uint4*)(arow + base + 16);
        }
    }

    float mst[8], sst[8];
    #pragma unroll
    for (int v = 0; v < 8; ++v) { mst[v] = -1e30f; sst[v] = 0.f; }

    // per-lane DMA addresses: each thread owns a 16 B slice, stride 2048 B
    const uint32_t lds0  = ((uint32_t)(uintptr_t)&lds[0][0]) + tid * 16;
    const uint64_t gbase = (uint64_t)(uintptr_t)Bm + tid * 16;

    // prologue: async-fill tiles 0 and 1
    async_fill_tile(lds0,              gbase);
    async_fill_tile(lds0 + TILE_BYTES, gbase + TILE_BYTES);
    wait_async_le8();          // tile 0 retired (in-order completion)
    __syncthreads();

    // per-lane base offset into an LDS stage for B-fragment gathers:
    // sub-tile n, step k -> (n*16 + lo)*DD + k*32 + hi*8 (elems 8..15 at +16)
    const int bbase = lo * DD + hi * 8;

    for (int t = 0; t < NITER; ++t) {
        // async DMA of tile t+2 into ring slot (t+2)&3 (2 chains of slack)
        const int t2 = t + 2;
        if (t2 < NITER)
            async_fill_tile(lds0 + (uint32_t)(t2 & (NBUF - 1)) * TILE_BYTES,
                            gbase + (uint64_t)t2 * TILE_BYTES);

        // preload all 16 B fragments (32 x ds_load_b128 burst)
        const uint16_t* sb = &lds[t & (NBUF - 1)][0];
        Frag16 bf[2][8];
        #pragma unroll
        for (int k = 0; k < 8; ++k) {
            const uint4* p0 = (const uint4*)(sb + bbase + k * 32);
            const uint4* p1 = (const uint4*)(sb + bbase + 16 * DD + k * 32);
            bf[0][k].q[0] = p0[0]; bf[0][k].q[1] = p0[2];
            bf[1][k].q[0] = p1[0]; bf[1][k].q[1] = p1[2];
        }

        // 16 WMMAs over K = 256, two column sub-tiles sharing each A fragment
        v8f acc0 = {}, acc1 = {};
        #pragma unroll
        for (int k = 0; k < 8; ++k) {
            acc0 = __builtin_amdgcn_wmma_f32_16x16x32_bf16(
                       false, afrag[k].v, false, bf[0][k].v, (short)0, acc0, false, false);
            acc1 = __builtin_amdgcn_wmma_f32_16x16x32_bf16(
                       false, afrag[k].v, false, bf[1][k].v, (short)0, acc1, false, false);
        }

        // online logsumexp update over 32 new columns; C/D layout: lane holds
        // col (lane&15) of rows v + 8*hi in acc[v]
        #pragma unroll
        for (int v = 0; v < 8; ++v) {
            const float x0 = acc0[v];
            const float x1 = acc1[v];
            const float mo = mst[v];
            const float mn = fmaxf(mo, fmaxf(x0, x1));
            sst[v] = sst[v] * __expf(mo - mn) + __expf(x0 - mn) + __expf(x1 - mn);
            mst[v] = mn;
        }

        // tile t+1 must be LDS-visible to all waves before next iteration
        if (t2 < NITER) wait_async_le8();   // older fill retired, newest in flight
        else            wait_async_zero();  // final fill: hard wait
        __syncthreads();
    }

    // merge (m,s) across the 16 lanes of each half (rows differ between halves)
    #pragma unroll
    for (int v = 0; v < 8; ++v) {
        float m = mst[v], s = sst[v];
        #pragma unroll
        for (int msk = 1; msk <= 8; msk <<= 1) {
            const float m2 = __shfl_xor(m, msk, 32);
            const float s2 = __shfl_xor(s, msk, 32);
            const float mn = fmaxf(m, m2);
            s = s * __expf(m - mn) + s2 * __expf(m2 - mn);
            m = mn;
        }
        mst[v] = m; sst[v] = s;
    }
    if (lo == 0) {
        #pragma unroll
        for (int v = 0; v < 8; ++v)
            lse[row0 + hi * 8 + v] = mst[v] + __logf(sst[v]);
    }
}

// ---------------------------------------------------------------------------
// 4) loss = mean_i( 0.5*(row_lse + col_lse) - diag )
// ---------------------------------------------------------------------------
__global__ void k_final(const float* __restrict__ rl, const float* __restrict__ cl,
                        const float* __restrict__ dg, float* __restrict__ out) {
    __shared__ float red[256];
    float acc = 0.f;
    for (int i = threadIdx.x; i < BN; i += 256)
        acc += 0.5f * (rl[i] + cl[i]) - dg[i];
    red[threadIdx.x] = acc;
    __syncthreads();
    for (int s = 128; s > 0; s >>= 1) {
        if (threadIdx.x < s) red[threadIdx.x] += red[threadIdx.x + s];
        __syncthreads();
    }
    if (threadIdx.x == 0) out[0] = red[0] / (float)BN;
}

// ---------------------------------------------------------------------------
extern "C" void kernel_launch(void* const* d_in, const int* in_sizes, int n_in,
                              void* d_out, int out_size, void* d_ws, size_t ws_size,
                              hipStream_t stream) {
    const float* M  = (const float*)d_in[0];
    const float* S  = (const float*)d_in[1];
    const float* sc = (const float*)d_in[2];

    char* ws = (char*)d_ws;
    uint16_t* Mbf = (uint16_t*)ws;                                   // 8 MB
    uint16_t* Sbf = (uint16_t*)(ws + (size_t)BN * DD * 2);           // 8 MB
    float* diag   = (float*)(ws + (size_t)BN * DD * 4);              // 64 KB
    float* rl     = diag + BN;                                       // 64 KB
    float* cl     = rl + BN;                                         // 64 KB
    float* out    = (float*)d_out;

    const int nconv_threads = BN * DD / 4;
    k_convert<<<nconv_threads / 256, 256, 0, stream>>>(M, S, sc, Mbf, Sbf);
    k_diag   <<<BN / 8,   256, 0, stream>>>(M, S, sc, diag);
    k_lse    <<<BN / 64,  128, 0, stream>>>(Mbf, Sbf, rl);   // modality -> sequence
    k_lse    <<<BN / 64,  128, 0, stream>>>(Sbf, Mbf, cl);   // sequence -> modality
    k_final  <<<1,        256, 0, stream>>>(rl, cl, diag, out);
}
